// WaveNetVQVAE_24352464568645
// MI455X (gfx1250) — matrix operand order repaired and checked by reading, not compiled, not run on hardware
//
#include <hip/hip_runtime.h>
#include <hip/hip_bf16.h>

typedef __attribute__((ext_vector_type(16))) __bf16       v16bf;
typedef __attribute__((ext_vector_type(8)))  float        v8f;
typedef __attribute__((ext_vector_type(8)))  unsigned int v8u;

#define B_    32
#define D_    64
#define T_    4096
#define K_    512
#define NROW  (B_ * T_)        // 131072 rows of flat
#define NEL   (B_ * D_ * T_)   // 8388608 elements
#define WAVES_PER_WG 4
#define NWG   512
#define NTILES (NROW / 16)                 // 8192 16-row tiles
#define WSTRIDE (NWG * WAVES_PER_WG)       // 2048 waves -> 4 tiles per wave

__device__ __forceinline__ unsigned short f2bf(float f) {
    // round-to-nearest-even f32 -> bf16
    unsigned u = __builtin_bit_cast(unsigned, f);
    unsigned r = u + 0x7FFFu + ((u >> 16) & 1u);
    return (unsigned short)(r >> 16);
}

__global__ void vq_init(float* loss_acc, unsigned int* hist) {
    int i = threadIdx.x;
    if (i == 0) *loss_acc = 0.0f;
    if (i < K_) hist[i] = 0u;
}

__global__ __launch_bounds__(128) void vq_main(
        const float* __restrict__ z,       // [B, D, T]
        const float* __restrict__ cb,      // [K, D]
        float* __restrict__ out,           // out[0]=loss, out+1 = quantized [B,D,T]
        float* __restrict__ loss_acc,      // ws
        unsigned int* __restrict__ hist) { // ws
    __shared__ unsigned short cb_bf[K_ * D_];        // 64 KB bf16 codebook, [n][d]
    __shared__ float          esq[K_];               // ||e||^2
    __shared__ int            widx[WAVES_PER_WG][16];

    const int tid = threadIdx.x;

    // ---- stage codebook -> bf16 LDS (coalesced), and e_sq (amortized over 4 tiles/wave)
    for (int i = tid; i < K_ * D_; i += 128) cb_bf[i] = f2bf(cb[i]);
    for (int n = tid; n < K_; n += 128) {
        float s = 0.0f;
        #pragma unroll 8
        for (int d = 0; d < D_; ++d) { float v = cb[n * D_ + d]; s += v * v; }
        esq[n] = s;
    }
    __syncthreads();

    const int wave  = tid >> 5;
    const int lane  = tid & 31;
    const int gwave = blockIdx.x * WAVES_PER_WG + wave;
    const int half  = lane >> 4;  // K-half split per 16-bit A/B layouts
    const int l15   = lane & 15;
    const unsigned int* cb_u = (const unsigned int*)cb_bf;
    float* qout = out + 1;

    for (int tile = gwave; tile < NTILES; tile += WSTRIDE) {   // uniform: 4 iters/thread
        const int b  = tile >> 8;                 // 256 tiles per batch (T=4096)
        const int t0 = (tile << 4) & (T_ - 1);
        const long zbase = (long)b * (D_ * T_);

        // ---- prefetch next tile's z rows (global_prefetch_b8), overlaps WMMA loop
        {
            const int nt2 = tile + WSTRIDE;
            if (nt2 < NTILES) {
                const long zb2 = (long)(nt2 >> 8) * (D_ * T_);
                const int  tp  = ((nt2 << 4) & (T_ - 1));
                __builtin_prefetch(&z[zb2 + (long)lane        * T_ + tp], 0, 3);
                __builtin_prefetch(&z[zb2 + (long)(lane + 32) * T_ + tp], 0, 3);
            }
        }

        // ---- load A tile (16 rows x K=64) pre-scaled by -2, in 16-bit A layout ----
        v8u apack[2];
        {
            const int t = t0 + l15;  // both lane-halves carry rows M=0..15
            #pragma unroll
            for (int kb = 0; kb < 2; ++kb) {
                #pragma unroll
                for (int v = 0; v < 8; ++v) {
                    const int kbase = 2 * v + 16 * half;
                    const int d0 = kb * 32 + kbase;
                    const float f0 = -2.0f * z[zbase + (long)d0 * T_ + t];
                    const float f1 = -2.0f * z[zbase + (long)(d0 + 1) * T_ + t];
                    apack[kb][v] = (unsigned)f2bf(f0) | ((unsigned)f2bf(f1) << 16);
                }
            }
        }
        const v16bf a0 = __builtin_bit_cast(v16bf, apack[0]);
        const v16bf a1 = __builtin_bit_cast(v16bf, apack[1]);

        float minv[8];
        int   mini[8];
        #pragma unroll
        for (int v = 0; v < 8; ++v) { minv[v] = 3.0e38f; mini[v] = 0; }

        // ---- distance GEMM: 32 N-tiles of 16 codes, K=64 via 2 chained bf16 WMMAs
        for (int nt = 0; nt < K_ / 16; ++nt) {
            const int n = nt * 16 + l15;          // this lane's code column
            const float e = esq[n];
            v8f c;
            #pragma unroll
            for (int v = 0; v < 8; ++v) c[v] = e; // C init = ||e_n||^2

            // B tiles from LDS: dword v = cb_u[n*32 + kb*16 + half*8 + v]
            const int base0 = n * 32 + half * 8;
            v8u bp0, bp1;
            #pragma unroll
            for (int v = 0; v < 8; ++v) bp0[v] = cb_u[base0 + v];
            #pragma unroll
            for (int v = 0; v < 8; ++v) bp1[v] = cb_u[base0 + 16 + v];
            const v16bf bb0 = __builtin_bit_cast(v16bf, bp0);
            const v16bf bb1 = __builtin_bit_cast(v16bf, bp1);

            c = __builtin_amdgcn_wmma_f32_16x16x32_bf16(false, a0, false, bb0,
                                                        (short)0, c, false, false);
            c = __builtin_amdgcn_wmma_f32_16x16x32_bf16(false, a1, false, bb1,
                                                        (short)0, c, false, false);

            // c[v] = ||e_n||^2 - 2 x_m . e_n  for row m = v + 8*half, col n
            #pragma unroll
            for (int v = 0; v < 8; ++v) {
                if (c[v] < minv[v]) { minv[v] = c[v]; mini[v] = n; }
            }
        }

        // ---- cross-lane argmin per row (branch-free: bitwise cond + cndmask) ----
        #pragma unroll
        for (int v = 0; v < 8; ++v) {
            float mv = minv[v];
            int   mi = mini[v];
            #pragma unroll
            for (int off = 8; off >= 1; off >>= 1) {
                const float ov = __shfl_xor(mv, off, 32);
                const int   oi = __shfl_xor(mi, off, 32);
                const int take = (int)(ov < mv) | ((int)(ov == mv) & (int)(oi < mi));
                mv = take ? ov : mv;
                mi = take ? oi : mi;
            }
            if (l15 == v) {                  // lanes v and v+16 own rows v, v+8
                const int m = v + 8 * half;
                widx[wave][m] = mi;
                atomicAdd(&hist[mi], 1u);
            }
        }
        __syncthreads();  // uniform; also orders widx store->load for the compiler

        // ---- gather + coalesced store of quantized, accumulate loss ----
        float acc = 0.0f;
        for (int e = 0; e < 32; ++e) {
            const int id   = e * 32 + lane;  // 1024 = 16 t x 64 d elements
            const int toff = id & 15;
            const int d    = id >> 4;
            const int idx  = widx[wave][toff];
            const float q  = cb[idx * D_ + d];     // f32 codebook, L2-resident
            const long off = zbase + (long)d * T_ + (t0 + toff);
            const float zz = z[off];
            const float df = q - zz;
            acc += df * df;
            qout[off] = q;
        }
        #pragma unroll
        for (int off = 16; off >= 1; off >>= 1) acc += __shfl_xor(acc, off, 32);
        if (lane == 0) atomicAdd(loss_acc, acc);
        // widx[wave] is wave-private and LDS is in-order per wave: no 2nd barrier
    }
}

__global__ __launch_bounds__(512) void vq_final(const float* loss_acc,
                                                const unsigned int* hist,
                                                float* out) {
    __shared__ float red[512];
    const int i = threadIdx.x;
    const float p = (float)hist[i] * (1.0f / (float)NROW);
    red[i] = p * logf(p + 1e-10f);
    __syncthreads();
    for (int s = 256; s > 0; s >>= 1) {
        if (i < s) red[i] += red[i + s];
        __syncthreads();
    }
    if (i == 0) {
        out[0]       = 1.25f * loss_acc[0] * (1.0f / (float)NEL);
        out[1 + NEL] = expf(-red[0]);
    }
}

extern "C" void kernel_launch(void* const* d_in, const int* in_sizes, int n_in,
                              void* d_out, int out_size, void* d_ws, size_t ws_size,
                              hipStream_t stream) {
    const float* z  = (const float*)d_in[0];   // [32, 64, 4096] f32
    const float* cb = (const float*)d_in[1];   // [512, 64] f32
    float* out = (float*)d_out;                // [loss, quantized(8388608), perplexity]

    float* loss_acc    = (float*)d_ws;
    unsigned int* hist = (unsigned int*)((char*)d_ws + 64);

    vq_init<<<1, 512, 0, stream>>>(loss_acc, hist);
    vq_main<<<NWG, 128, 0, stream>>>(z, cb, out, loss_acc, hist);
    vq_final<<<1, 512, 0, stream>>>(loss_acc, hist, out);
}
